// MoD_45183055953977
// MI455X (gfx1250) — matrix operand order repaired and checked by reading, not compiled, not run on hardware
//
#include <hip/hip_runtime.h>
#include <math.h>

#define BB   4
#define SS   4096
#define DD   2048
#define DFFN 8192
#define KSEL 512
#define ROWS (BB * KSEL)   // 2048 gathered rows total

typedef __attribute__((ext_vector_type(16))) __bf16 v16bf;
typedef __attribute__((ext_vector_type(8)))  float  v8f;
typedef unsigned int u32x4 __attribute__((ext_vector_type(4)));
typedef int          i32x4 __attribute__((ext_vector_type(4)));
typedef int          i32x8 __attribute__((ext_vector_type(8)));

// fp32 -> (hi, lo) bf16 split: hi = bf16(x), lo = bf16(x - hi)
__device__ __forceinline__ void bsplit(float f, __bf16 &hi, __bf16 &lo) {
  __bf16 h = (__bf16)f;
  hi = h;
  lo = (__bf16)(f - (float)h);
}

// ---------------------------------------------------------------------------
// TDM: 2D bf16 tile load (tile_d1 rows x tile_d0 elems) global -> LDS, with
// LDS padding of 4 DWORDs every 16 DWORDs (64B row -> 80B stride == [.][40]).
// Descriptor packing per CDNA5 ISA.
// ---------------------------------------------------------------------------
__device__ __forceinline__ void tdm_load_2d_bf16(const void* gaddr,
                                                 unsigned lds_byte_addr,
                                                 unsigned tensor_d0,
                                                 unsigned tensor_d1,
                                                 unsigned stride_d0,
                                                 unsigned tile_d0,
                                                 unsigned tile_d1) {
  unsigned long long ga = (unsigned long long)(uintptr_t)gaddr;
  u32x4 g0;
  g0[0] = 1u;                                        // count=1, user descriptor
  g0[1] = lds_byte_addr;                             // LDS dest (bytes)
  g0[2] = (unsigned)(ga & 0xFFFFFFFFu);              // global_addr[31:0]
  g0[3] = (unsigned)((ga >> 32) & 0x01FFFFFFu)       // global_addr[56:32]
        | (2u << 30);                                // type = 2 (image)
  i32x8 g1;
  g1[0] = (int)((1u << 16)                           // data_size = 2 bytes
              | (1u << 20)                           // pad_enable
              | (3u << 22)                           // pad_interval: 16 DWORDs
              | (3u << 25));                         // pad_amount: 4 DWORDs
  g1[1] = (int)((tensor_d0 & 0xFFFFu) << 16);        // tensor_dim0[15:0]
  g1[2] = (int)(((tensor_d0 >> 16) & 0xFFFFu)        // tensor_dim0[31:16]
              | ((tensor_d1 & 0xFFFFu) << 16));      // tensor_dim1[15:0]
  g1[3] = (int)(((tensor_d1 >> 16) & 0xFFFFu)        // tensor_dim1[31:16]
              | ((tile_d0 & 0xFFFFu) << 16));        // tile_dim0
  g1[4] = (int)(tile_d1 & 0xFFFFu);                  // tile_dim1 (tile_dim2=0)
  g1[5] = (int)stride_d0;                            // tensor_dim0_stride[31:0]
  g1[6] = 0;                                         // stride hi / dim1_stride lo
  g1[7] = 0;
  i32x4 z4; z4[0] = 0; z4[1] = 0; z4[2] = 0; z4[3] = 0;
#if defined(__clang_major__) && (__clang_major__ >= 23)
  i32x8 z8; z8[0]=0; z8[1]=0; z8[2]=0; z8[3]=0; z8[4]=0; z8[5]=0; z8[6]=0; z8[7]=0;
  __builtin_amdgcn_tensor_load_to_lds(g0, g1, z4, z4, z8, 0);
#else
  __builtin_amdgcn_tensor_load_to_lds(g0, g1, z4, z4, 0);
#endif
}

// ---------------------------------------------------------------- router GEMV
__global__ __launch_bounds__(256) void mod_router(const float* __restrict__ x,
                                                  const float* __restrict__ Wr,
                                                  const float* __restrict__ br,
                                                  float* __restrict__ logits) {
  __shared__ float red[256];
  const int token = blockIdx.x;                 // 0 .. B*S-1
  const float* xr = x + (size_t)token * DD;
  float s = 0.f;
  for (int i = threadIdx.x; i < DD; i += 256) s = fmaf(xr[i], Wr[i], s);
  red[threadIdx.x] = s;
  __syncthreads();
  for (int off = 128; off > 0; off >>= 1) {
    if (threadIdx.x < off) red[threadIdx.x] += red[threadIdx.x + off];
    __syncthreads();
  }
  if (threadIdx.x == 0) logits[token] = red[0] + br[0];
}

// ------------------------------------------- per-batch top-k + sort + softmax
__global__ __launch_bounds__(1024) void mod_topk(const float* __restrict__ logits,
                                                 int* __restrict__ tokens,
                                                 float* __restrict__ rw) {
  __shared__ float val[SS];
  __shared__ int   idx[SS];
  __shared__ float red[1024];
  const int b = blockIdx.x, tid = threadIdx.x;

  for (int i = tid; i < SS; i += 1024) { val[i] = logits[b * SS + i]; idx[i] = i; }
  __syncthreads();

  // Bitonic sort: larger value first; ties -> lower index first (== jax top_k)
  for (int k = 2; k <= SS; k <<= 1) {
    for (int j = k >> 1; j > 0; j >>= 1) {
      for (int i = tid; i < SS; i += 1024) {
        int p = i ^ j;
        if (p > i) {
          float v0 = val[i], v1 = val[p];
          int   i0 = idx[i], i1 = idx[p];
          bool before01 = (v0 > v1) || (v0 == v1 && i0 < i1);
          bool up = ((i & k) == 0);
          if (up ? !before01 : before01) {
            val[i] = v1; val[p] = v0; idx[i] = i1; idx[p] = i0;
          }
        }
      }
      __syncthreads();
    }
  }

  // softmax over the 512 winners (max == val[0] after descending sort)
  float m = val[0];
  float ls = 0.f;
  for (int i = tid; i < KSEL; i += 1024) ls += expf(val[i] - m);
  red[tid] = ls;
  __syncthreads();
  for (int off = 512; off > 0; off >>= 1) {
    if (tid < off) red[tid] += red[tid + off];
    __syncthreads();
  }
  float inv = 1.f / red[0];
  __syncthreads();
  for (int i = tid; i < KSEL; i += 1024) val[i] = expf(val[i] - m) * inv;
  __syncthreads();

  // Sort the 512 (token, weight) pairs by token index ascending (== argsort)
  for (int k = 2; k <= KSEL; k <<= 1) {
    for (int j = k >> 1; j > 0; j >>= 1) {
      for (int i = tid; i < KSEL; i += 1024) {
        int p = i ^ j;
        if (p > i) {
          int i0 = idx[i], i1 = idx[p];
          bool up = ((i & k) == 0);
          bool before01 = i0 < i1;
          if (up ? !before01 : before01) {
            float v0 = val[i]; val[i] = val[p]; val[p] = v0;
            idx[i] = i1; idx[p] = i0;
          }
        }
      }
      __syncthreads();
    }
  }
  for (int i = tid; i < KSEL; i += 1024) {
    tokens[b * KSEL + i] = idx[i];
    rw[b * KSEL + i]     = val[i];
  }
}

// ------------------------------------------------------------ GEMM1 + GELU
// A = gathered x rows [2048 x 2048], B = W1 [2048 x 8192].
// bf16x3 split, 128x128x32 block tile, 8 waves (2x4), 4x2 16x16 tiles/wave.
// Epilogue stores h pre-split as bf16 hi/lo planes (feeds TDM in GEMM2).
__global__ __launch_bounds__(256) void mod_ffn1(const float* __restrict__ x,
                                                const float* __restrict__ W1,
                                                const float* __restrict__ b1,
                                                const int* __restrict__ tokens,
                                                __bf16* __restrict__ hhi,
                                                __bf16* __restrict__ hlo) {
  __shared__ __align__(16) __bf16 Ahi[128][40];
  __shared__ __align__(16) __bf16 Alo[128][40];
  __shared__ __align__(16) __bf16 Bhi[128][40];   // stored transposed: [n][k]
  __shared__ __align__(16) __bf16 Blo[128][40];
  __shared__ size_t rowbase[128];

  const int tid  = threadIdx.x;
  const int lane = tid & 31, wave = tid >> 5;
  const int wm = wave >> 2, wn = wave & 3;        // 2 x 4 wave grid
  const int lr = lane & 15, lh = lane >> 4;
  const int bm = blockIdx.y * 128;
  const int bn = blockIdx.x * 128;

  if (tid < 128) {
    int rg = bm + tid;
    rowbase[tid] = ((size_t)(rg >> 9) * SS + (size_t)tokens[rg]) * (size_t)DD;
  }

  v8f acc[4][2];
  #pragma unroll
  for (int mt = 0; mt < 4; ++mt)
    #pragma unroll
    for (int nt = 0; nt < 2; ++nt)
      #pragma unroll
      for (int e = 0; e < 8; ++e) acc[mt][nt][e] = 0.f;

  for (int kc = 0; kc < DD; kc += 32) {
    __syncthreads();
    // stage A (gathered) tile, split to hi/lo bf16
    #pragma unroll
    for (int i = 0; i < 4; ++i) {
      int f = tid + i * 256;
      int r = f >> 3, kq = f & 7;
      float4 w = *(const float4*)(x + rowbase[r] + (size_t)(kc + kq * 4));
      const float* wp = (const float*)&w;
      #pragma unroll
      for (int j = 0; j < 4; ++j) {
        __bf16 hi, lo; bsplit(wp[j], hi, lo);
        Ahi[r][kq * 4 + j] = hi; Alo[r][kq * 4 + j] = lo;
      }
    }
    // stage B tile transposed, split to hi/lo bf16
    #pragma unroll
    for (int i = 0; i < 4; ++i) {
      int f = tid + i * 256;
      int kr = f >> 5, nq = f & 31;
      float4 w = *(const float4*)(W1 + (size_t)(kc + kr) * DFFN + bn + nq * 4);
      const float* wp = (const float*)&w;
      #pragma unroll
      for (int j = 0; j < 4; ++j) {
        __bf16 hi, lo; bsplit(wp[j], hi, lo);
        Bhi[nq * 4 + j][kr] = hi; Blo[nq * 4 + j][kr] = lo;
      }
    }
    __syncthreads();

    union { v16bf v; uint4 q[2]; } bh[2], bl[2];
    #pragma unroll
    for (int nt = 0; nt < 2; ++nt) {
      int n = wn * 32 + nt * 16 + lr;
      int kb = lh * 8;   // lane<16 -> K {0..7,16..23}, lane>=16 -> {8..15,24..31}
      bh[nt].q[0] = *reinterpret_cast<const uint4*>(&Bhi[n][kb]);
      bh[nt].q[1] = *reinterpret_cast<const uint4*>(&Bhi[n][kb + 16]);
      bl[nt].q[0] = *reinterpret_cast<const uint4*>(&Blo[n][kb]);
      bl[nt].q[1] = *reinterpret_cast<const uint4*>(&Blo[n][kb + 16]);
    }
    #pragma unroll
    for (int mt = 0; mt < 4; ++mt) {
      union { v16bf v; uint4 q[2]; } ah, al;
      int m = wm * 64 + mt * 16 + lr;
      int kb = lh * 8;   // ISA A layout: lane<16 -> K {0..7,16..23}, lane>=16 -> {8..15,24..31}
      ah.q[0] = *reinterpret_cast<const uint4*>(&Ahi[m][kb]);
      ah.q[1] = *reinterpret_cast<const uint4*>(&Ahi[m][16 + kb]);
      al.q[0] = *reinterpret_cast<const uint4*>(&Alo[m][kb]);
      al.q[1] = *reinterpret_cast<const uint4*>(&Alo[m][16 + kb]);
      #pragma unroll
      for (int nt = 0; nt < 2; ++nt) {
        acc[mt][nt] = __builtin_amdgcn_wmma_f32_16x16x32_bf16(false, ah.v, false, bh[nt].v, (short)0, acc[mt][nt], false, false);
        acc[mt][nt] = __builtin_amdgcn_wmma_f32_16x16x32_bf16(false, ah.v, false, bl[nt].v, (short)0, acc[mt][nt], false, false);
        acc[mt][nt] = __builtin_amdgcn_wmma_f32_16x16x32_bf16(false, al.v, false, bh[nt].v, (short)0, acc[mt][nt], false, false);
      }
    }
  }

  // epilogue: exact GELU, store h as split bf16 hi/lo planes
  #pragma unroll
  for (int mt = 0; mt < 4; ++mt)
    #pragma unroll
    for (int nt = 0; nt < 2; ++nt) {
      int mg = bm + wm * 64 + mt * 16 + lh * 8;
      int ng = bn + wn * 32 + nt * 16 + lr;
      float bias = b1[ng];
      #pragma unroll
      for (int r = 0; r < 8; ++r) {
        float v = acc[mt][nt][r] + bias;
        v = 0.5f * v * (1.f + erff(v * 0.70710678118654752f));
        __bf16 hi, lo; bsplit(v, hi, lo);
        size_t o = (size_t)(mg + r) * DFFN + ng;
        hhi[o] = hi;
        hlo[o] = lo;
      }
    }
}

// --------------------------------------- GEMM2 + weight-scale + scatter-add
// A = pre-split h (bf16 hi/lo) -> staged by the Tensor Data Mover with
// hardware LDS padding; B = W2 split on the fly on the vector path.
__global__ __launch_bounds__(256) void mod_ffn2(const __bf16* __restrict__ hhi,
                                                const __bf16* __restrict__ hlo,
                                                const float* __restrict__ W2,
                                                const float* __restrict__ b2,
                                                const int* __restrict__ tokens,
                                                const float* __restrict__ rw,
                                                float* __restrict__ out) {
  __shared__ __align__(16) __bf16 Ahi[128][40];   // filled by TDM (padded rows)
  __shared__ __align__(16) __bf16 Alo[128][40];
  __shared__ __align__(16) __bf16 Bhi[128][40];
  __shared__ __align__(16) __bf16 Blo[128][40];
  __shared__ int   rtok[128];
  __shared__ float rrw[128];

  const int tid  = threadIdx.x;
  const int lane = tid & 31, wave = tid >> 5;
  const int wm = wave >> 2, wn = wave & 3;
  const int lr = lane & 15, lh = lane >> 4;
  const int bm = blockIdx.y * 128;
  const int bn = blockIdx.x * 128;

  if (tid < 128) {
    int rg = bm + tid;
    rtok[tid] = tokens[rg];
    rrw[tid]  = rw[rg];
  }

  // LDS byte addresses of the TDM destinations (flat LDS addr, low 32 bits)
  const unsigned ldsAhi = (unsigned)(uintptr_t)(&Ahi[0][0]);
  const unsigned ldsAlo = (unsigned)(uintptr_t)(&Alo[0][0]);

  v8f acc[4][2];
  #pragma unroll
  for (int mt = 0; mt < 4; ++mt)
    #pragma unroll
    for (int nt = 0; nt < 2; ++nt)
      #pragma unroll
      for (int e = 0; e < 8; ++e) acc[mt][nt][e] = 0.f;

  for (int kc = 0; kc < DFFN; kc += 32) {
    __syncthreads();
    // TDM: DMA the 128x32 bf16 A tiles (hi & lo) straight into padded LDS
    if (wave == 0) {
      tdm_load_2d_bf16(hhi + (size_t)bm * DFFN + kc, ldsAhi,
                       DFFN, ROWS, DFFN, 32, 128);
      tdm_load_2d_bf16(hlo + (size_t)bm * DFFN + kc, ldsAlo,
                       DFFN, ROWS, DFFN, 32, 128);
    }
    // Vector path: stage W2 tile transposed, split to hi/lo bf16 (overlaps TDM)
    #pragma unroll
    for (int i = 0; i < 4; ++i) {
      int f = tid + i * 256;
      int kr = f >> 5, nq = f & 31;
      float4 w = *(const float4*)(W2 + (size_t)(kc + kr) * DD + bn + nq * 4);
      const float* wp = (const float*)&w;
      #pragma unroll
      for (int j = 0; j < 4; ++j) {
        __bf16 hi, lo; bsplit(wp[j], hi, lo);
        Bhi[nq * 4 + j][kr] = hi; Blo[nq * 4 + j][kr] = lo;
      }
    }
    if (wave == 0) __builtin_amdgcn_s_wait_tensorcnt(0);
    __syncthreads();

    union { v16bf v; uint4 q[2]; } bh[2], bl[2];
    #pragma unroll
    for (int nt = 0; nt < 2; ++nt) {
      int n = wn * 32 + nt * 16 + lr;
      int kb = lh * 8;
      bh[nt].q[0] = *reinterpret_cast<const uint4*>(&Bhi[n][kb]);
      bh[nt].q[1] = *reinterpret_cast<const uint4*>(&Bhi[n][kb + 16]);
      bl[nt].q[0] = *reinterpret_cast<const uint4*>(&Blo[n][kb]);
      bl[nt].q[1] = *reinterpret_cast<const uint4*>(&Blo[n][kb + 16]);
    }
    #pragma unroll
    for (int mt = 0; mt < 4; ++mt) {
      union { v16bf v; uint4 q[2]; } ah, al;
      int m = wm * 64 + mt * 16 + lr;
      int kb = lh * 8;
      ah.q[0] = *reinterpret_cast<const uint4*>(&Ahi[m][kb]);
      ah.q[1] = *reinterpret_cast<const uint4*>(&Ahi[m][16 + kb]);
      al.q[0] = *reinterpret_cast<const uint4*>(&Alo[m][kb]);
      al.q[1] = *reinterpret_cast<const uint4*>(&Alo[m][16 + kb]);
      #pragma unroll
      for (int nt = 0; nt < 2; ++nt) {
        acc[mt][nt] = __builtin_amdgcn_wmma_f32_16x16x32_bf16(false, ah.v, false, bh[nt].v, (short)0, acc[mt][nt], false, false);
        acc[mt][nt] = __builtin_amdgcn_wmma_f32_16x16x32_bf16(false, ah.v, false, bl[nt].v, (short)0, acc[mt][nt], false, false);
        acc[mt][nt] = __builtin_amdgcn_wmma_f32_16x16x32_bf16(false, al.v, false, bh[nt].v, (short)0, acc[mt][nt], false, false);
      }
    }
  }

  // epilogue: out[b, token, n] += rw * (acc + b2[n]); rows unique -> race-free
  #pragma unroll
  for (int mt = 0; mt < 4; ++mt)
    #pragma unroll
    for (int nt = 0; nt < 2; ++nt) {
      int mlb = wm * 64 + mt * 16 + lh * 8;
      int ng  = bn + wn * 32 + nt * 16 + lr;
      float bias = b2[ng];
      #pragma unroll
      for (int r = 0; r < 8; ++r) {
        int ml = mlb + r;
        int rg = bm + ml;
        float v = rrw[ml] * (acc[mt][nt][r] + bias);
        float* p = out + ((size_t)(rg >> 9) * SS + (size_t)rtok[ml]) * DD + ng;
        *p += v;
      }
    }
}

extern "C" void kernel_launch(void* const* d_in, const int* in_sizes, int n_in,
                              void* d_out, int out_size, void* d_ws, size_t ws_size,
                              hipStream_t stream) {
  const float* x  = (const float*)d_in[0];
  const float* Wr = (const float*)d_in[1];
  const float* br = (const float*)d_in[2];
  const float* W1 = (const float*)d_in[3];
  const float* b1 = (const float*)d_in[4];
  const float* W2 = (const float*)d_in[5];
  const float* b2 = (const float*)d_in[6];
  float* out = (float*)d_out;

  char* ws = (char*)d_ws;
  float*  logits = (float*)(ws);                          // 64 KiB
  int*    tokens = (int*)(ws + (64 << 10));               // 8 KiB
  float*  rw     = (float*)(ws + (72 << 10));             // 8 KiB
  __bf16* hhi    = (__bf16*)(ws + (128 << 10));           // 32 MiB
  __bf16* hlo    = (__bf16*)(ws + (128 << 10) + (size_t)ROWS * DFFN * 2);  // 32 MiB

  mod_router<<<dim3(BB * SS), dim3(256), 0, stream>>>(x, Wr, br, logits);
  mod_topk<<<dim3(BB), dim3(1024), 0, stream>>>(logits, tokens, rw);
  hipMemcpyAsync(out, x, (size_t)BB * SS * DD * sizeof(float),
                 hipMemcpyDeviceToDevice, stream);
  mod_ffn1<<<dim3(DFFN / 128, ROWS / 128), dim3(256), 0, stream>>>(x, W1, b1, tokens, hhi, hlo);
  mod_ffn2<<<dim3(DD / 128, ROWS / 128), dim3(256), 0, stream>>>(hhi, hlo, W2, b2, tokens, rw, out);
}
